// LSTMBase_81853486727941
// MI455X (gfx1250) — matrix operand order repaired and checked by reading, not compiled, not run on hardware
//
#include <hip/hip_runtime.h>

typedef __attribute__((ext_vector_type(8)))  __bf16 v8bf;
typedef __attribute__((ext_vector_type(16))) __bf16 v16bf;
typedef __attribute__((ext_vector_type(8)))  float  v8f;

#define HID   1024
#define TSEQ  512
#define BATCH 64

__global__ void f32_to_bf16_k(const float* __restrict__ src, __bf16* __restrict__ dst, size_t n) {
    size_t i = (size_t)blockIdx.x * blockDim.x + threadIdx.x;
    size_t stride = (size_t)gridDim.x * blockDim.x;
    for (; i < n; i += stride) dst[i] = (__bf16)src[i];
}

__device__ __forceinline__ float sigmoidf_(float x) { return 1.0f / (1.0f + __expf(-x)); }

__device__ __forceinline__ v16bf cat16(const __bf16* lo, const __bf16* hi) {
    v8bf a = *(const v8bf*)lo;
    v8bf b = *(const v8bf*)hi;
    return __builtin_shufflevector(a, b, 0,1,2,3,4,5,6,7,8,9,10,11,12,13,14,15);
}

#define WMMA_BF16(A, B, C) \
    __builtin_amdgcn_wmma_f32_16x16x32_bf16(false, (A), false, (B), (short)0, (C), false, false)

// One timestep of the LSTM recurrence, fully fused:
//   gates = [x_t ; h_{t-1}] @ [Wx ; Wh]^T   (K = 2048, bf16 WMMA, f32 accum)
//   c_t = sigmoid(f)*c + sigmoid(i)*tanh(g);  h_t = sigmoid(o)*tanh(c_t)
//
// Grid: 64 blocks  -> hidden slice n0 = blockIdx.x*16 (16 columns, all 4 gates).
// Block: 64 threads (2 waves). Wave w owns batch rows [w*32, w*32+32) as two
// 16-row M-tiles and ALL FOUR gates (8 f32 accumulators). Because the WMMA C/D
// layout puts element rr of every gate accumulator at the same (row, col), the
// whole cell update happens in registers: no LDS exchange, no barrier.
__global__ __launch_bounds__(64)
void lstm_step_k(int t,
                 const __bf16* __restrict__ X,    // [B][T][H] bf16
                 const __bf16* __restrict__ Wx,   // [4][H][H] bf16 (i,f,g,o)
                 const __bf16* __restrict__ Wh,   // [4][H][H] bf16 (i,f,g,o)
                 const __bf16* __restrict__ h_in, // [B][H] bf16
                 __bf16*       __restrict__ h_out,
                 float*        __restrict__ c,    // [B][H] f32 (persistent)
                 const float* __restrict__ b_ii, const float* __restrict__ b_hi,
                 const float* __restrict__ b_if, const float* __restrict__ b_hf,
                 const float* __restrict__ b_ig, const float* __restrict__ b_hg,
                 const float* __restrict__ b_io, const float* __restrict__ b_ho,
                 float* __restrict__ out)         // [B][T][H] f32, then [B][H] h_T
{
    const int tid  = threadIdx.x;
    const int w    = tid >> 5;        // wave 0..1
    const int lane = tid & 31;
    const int lr   = lane & 15;
    const int hi   = lane >> 4;
    const int n0   = blockIdx.x * 16;
    const int n    = n0 + lr;         // hidden column / W row (lane-invariant over rr/mt)
    const int r0   = w * 32 + lr;     // first batch row; second M-tile at r0+16

    v8f acc[2][4] = {};               // [m-tile][gate], fully register-resident

    const __bf16* ax0 = X + ((size_t)r0 * TSEQ + t) * HID;
    const __bf16* ax1 = ax0 + (size_t)16 * TSEQ * HID;
    const __bf16* ah0 = h_in + (size_t)r0 * HID;
    const __bf16* ah1 = ah0 + (size_t)16 * HID;
    const __bf16* wxp[4], * whp[4];
    #pragma unroll
    for (int z = 0; z < 4; ++z) {
        wxp[z] = Wx + ((size_t)z * HID + (size_t)n) * HID;
        whp[z] = Wh + ((size_t)z * HID + (size_t)n) * HID;
    }

    // ---- x_t contribution (K = 0..1023) ----
    #pragma unroll 2
    for (int k0 = 0; k0 < HID; k0 += 32) {
        // A frag: lanes 0-15 hold K {k0..k0+7, k0+16..k0+23}; lanes 16-31 the +8 halves
        v16bf a0 = cat16(ax0 + k0 + hi * 8, ax0 + k0 + 16 + hi * 8);
        v16bf a1 = cat16(ax1 + k0 + hi * 8, ax1 + k0 + 16 + hi * 8);
        #pragma unroll
        for (int z = 0; z < 4; ++z) {
            // B frag: lane = column n, K {k0+hi*8..+7, k0+16+hi*8..+7} from W row n (matches A, no transpose)
            v16bf b = cat16(wxp[z] + k0 + hi * 8, wxp[z] + k0 + 16 + hi * 8);
            acc[0][z] = WMMA_BF16(a0, b, acc[0][z]);
            acc[1][z] = WMMA_BF16(a1, b, acc[1][z]);
        }
    }

    // ---- h_{t-1} contribution (K = 1024..2047) ----
    #pragma unroll 2
    for (int k0 = 0; k0 < HID; k0 += 32) {
        v16bf a0 = cat16(ah0 + k0 + hi * 8, ah0 + k0 + 16 + hi * 8);
        v16bf a1 = cat16(ah1 + k0 + hi * 8, ah1 + k0 + 16 + hi * 8);
        #pragma unroll
        for (int z = 0; z < 4; ++z) {
            v16bf b = cat16(whp[z] + k0 + hi * 8, whp[z] + k0 + 16 + hi * 8);
            acc[0][z] = WMMA_BF16(a0, b, acc[0][z]);
            acc[1][z] = WMMA_BF16(a1, b, acc[1][z]);
        }
    }

    // ---- in-register cell update (all 4 gates live in this lane) ----
    const float bi = b_ii[n] + b_hi[n];
    const float bf = b_if[n] + b_hf[n];
    const float bg = b_ig[n] + b_hg[n];
    const float bo = b_io[n] + b_ho[n];

    #pragma unroll
    for (int mt = 0; mt < 2; ++mt) {
        #pragma unroll
        for (int rr = 0; rr < 8; ++rr) {
            // C/D layout: element rr -> row (hi*8 + rr) of this 16-row tile, column lr
            const int b = (w * 2 + mt) * 16 + hi * 8 + rr;
            const float it = sigmoidf_(acc[mt][0][rr] + bi);
            const float ft = sigmoidf_(acc[mt][1][rr] + bf);
            const float gt = tanhf    (acc[mt][2][rr] + bg);
            const float ot = sigmoidf_(acc[mt][3][rr] + bo);
            const size_t cidx = (size_t)b * HID + n;
            const float cn = ft * c[cidx] + it * gt;
            c[cidx] = cn;
            const float hn = ot * tanhf(cn);
            h_out[cidx] = (__bf16)hn;
            out[((size_t)b * TSEQ + t) * HID + n] = ot;          // stacked o_t, [B][T][H]
            if (t == TSEQ - 1)
                out[(size_t)BATCH * TSEQ * HID + cidx] = hn;     // h_T
        }
    }
}

extern "C" void kernel_launch(void* const* d_in, const int* in_sizes, int n_in,
                              void* d_out, int out_size, void* d_ws, size_t ws_size,
                              hipStream_t stream) {
    (void)in_sizes; (void)n_in; (void)out_size; (void)ws_size;
    const float* X    = (const float*)d_in[0];
    const float* W_ii = (const float*)d_in[1];  const float* b_ii = (const float*)d_in[2];
    const float* W_hi = (const float*)d_in[3];  const float* b_hi = (const float*)d_in[4];
    const float* W_if = (const float*)d_in[5];  const float* b_if = (const float*)d_in[6];
    const float* W_hf = (const float*)d_in[7];  const float* b_hf = (const float*)d_in[8];
    const float* W_ig = (const float*)d_in[9];  const float* b_ig = (const float*)d_in[10];
    const float* W_hg = (const float*)d_in[11]; const float* b_hg = (const float*)d_in[12];
    const float* W_io = (const float*)d_in[13]; const float* b_io = (const float*)d_in[14];
    const float* W_ho = (const float*)d_in[15]; const float* b_ho = (const float*)d_in[16];
    float* out = (float*)d_out;

    // Workspace layout (bf16 operands + recurrent state), ~80.5 MB total
    char* ws = (char*)d_ws;
    const size_t XN = (size_t)BATCH * TSEQ * HID;
    __bf16* Xbf  = (__bf16*)ws;  ws += XN * 2;
    __bf16* Wxbf = (__bf16*)ws;  ws += (size_t)4 * HID * HID * 2;
    __bf16* Whbf = (__bf16*)ws;  ws += (size_t)4 * HID * HID * 2;
    __bf16* h0   = (__bf16*)ws;  ws += (size_t)BATCH * HID * 2;
    __bf16* h1   = (__bf16*)ws;  ws += (size_t)BATCH * HID * 2;
    float*  c    = (float*)ws;

    hipMemsetAsync(h0, 0, (size_t)BATCH * HID * 2, stream);           // h_0 = 0
    hipMemsetAsync(c,  0, (size_t)BATCH * HID * 4, stream);           // c_0 = 0

    // fp32 -> bf16 conversions
    f32_to_bf16_k<<<2048, 256, 0, stream>>>(X, Xbf, XN);
    const float* wx[4] = {W_ii, W_if, W_ig, W_io};
    const float* wh[4] = {W_hi, W_hf, W_hg, W_ho};
    for (int z = 0; z < 4; ++z) {
        f32_to_bf16_k<<<512, 256, 0, stream>>>(wx[z], Wxbf + (size_t)z * HID * HID, (size_t)HID * HID);
        f32_to_bf16_k<<<512, 256, 0, stream>>>(wh[z], Whbf + (size_t)z * HID * HID, (size_t)HID * HID);
    }

    // Sequential recurrence, double-buffered h
    for (int t = 0; t < TSEQ; ++t) {
        const __bf16* hin = (t & 1) ? h1 : h0;
        __bf16*      hout = (t & 1) ? h0 : h1;
        lstm_step_k<<<64, 64, 0, stream>>>(t, Xbf, Wxbf, Whbf, hin, hout, c,
                                           b_ii, b_hi, b_if, b_hf,
                                           b_ig, b_hg, b_io, b_ho, out);
    }
}
